// Block_86096914416138
// MI455X (gfx1250) — matrix operand order repaired and checked by reading, not compiled, not run on hardware
//
#include <hip/hip_runtime.h>

typedef __attribute__((ext_vector_type(16))) _Float16 v16h;
typedef __attribute__((ext_vector_type(8)))  float    v8f;

#define Bb 16
#define Cc 1024
#define Tt 1024
#define NROWS (Bb * Cc)          /* 16384 independent (b,c) sequences */
#define NTOT  ((size_t)NROWS * Tt)

// ---------------------------------------------------------------------------
// Kernel 1: one wave per (b,c) row. Wave-parallel linear recurrence
//   m[t] = beta*m[t-1] + cur[t]   (Hillis-Steele with beta^offset multipliers)
// Writes membrane (slot 2) and spikes (temporarily into slot 0).
// ---------------------------------------------------------------------------
__global__ __launch_bounds__(256) void snn_scan_kernel(
    const float* __restrict__ current,
    const float* __restrict__ beta,
    const float* __restrict__ v_init,
    const float* __restrict__ v_th,
    float* __restrict__ spike_out,     // d_out slot 0 (temp)
    float* __restrict__ membrane_out)  // d_out slot 2
{
    const int lane = threadIdx.x & 31;
    const int row  = blockIdx.x * (blockDim.x >> 5) + (threadIdx.x >> 5);
    if (row >= NROWS) return;

    const float bet = beta[row & (Cc - 1)];
    const float vin = v_init[row];
    const float b2 = bet * bet, b4 = b2 * b2, b8 = b4 * b4, b16 = b8 * b8;

    // p = beta^(lane+1)  (multiplicative wave scan)
    float p = bet;
    {
        float u;
        u = __shfl_up(p, 1, 32);  if (lane >= 1)  p *= u;
        u = __shfl_up(p, 2, 32);  if (lane >= 2)  p *= u;
        u = __shfl_up(p, 4, 32);  if (lane >= 4)  p *= u;
        u = __shfl_up(p, 8, 32);  if (lane >= 8)  p *= u;
        u = __shfl_up(p, 16, 32); if (lane >= 16) p *= u;
    }

    const size_t base = (size_t)row * Tt;
    float carry = 0.0f;   // membrane at end of previous 32-chunk

    for (int ch = 0; ch < Tt / 32; ++ch) {
        const int t = ch * 32 + lane;
        float m = current[base + t];
        if (t == 0) m += bet * vin;      // fused v_init injection

        float u;
        u = __shfl_up(m, 1, 32);  if (lane >= 1)  m += bet * u;
        u = __shfl_up(m, 2, 32);  if (lane >= 2)  m += b2  * u;
        u = __shfl_up(m, 4, 32);  if (lane >= 4)  m += b4  * u;
        u = __shfl_up(m, 8, 32);  if (lane >= 8)  m += b8  * u;
        u = __shfl_up(m, 16, 32); if (lane >= 16) m += b16 * u;

        m += p * carry;                  // carry in: beta^(lane+1) * m[t0-1]
        carry = __shfl(m, 31, 32);

        membrane_out[base + t] = m;
        spike_out[base + t] = (m - v_th[base + t]) > 0.0f ? 1.0f : 0.0f;
    }
}

// ---------------------------------------------------------------------------
// Kernel 2: one wave per group of 16 rows. Double-cumsum of spikes computed
// as chunked matmul with the causal kernel W'[k][n] = (n-k+1), k<=n:
//   D(16 rows x 16 t) = A(spikes 16x32, K16..31=0) x B(W', const) + C(carries)
// Exact in f16xf16->f32: spikes in {0,1}, weights <= 16, z integer < 2^24.
// Carries: c1 = D[.,15]-D[.,14], c2 = D[.,15];  C_next[m][n] = (n+1)c1 + c2.
// out = (z==1) ? 1 : 0 overwrites the spike scratch (slot 0); z -> slot 1.
// ---------------------------------------------------------------------------
__global__ __launch_bounds__(256) void snn_phi_wmma_kernel(
    const float* spikes,   // d_out slot 0 (read)   -- aliases `out`
    float*       out,      // d_out slot 0 (write)
    float* __restrict__ z_out)  // d_out slot 1
{
    const int lane = threadIdx.x & 31;
    const int wave = blockIdx.x * (blockDim.x >> 5) + (threadIdx.x >> 5);
    const int rowbase = wave * 16;
    if (rowbase >= NROWS) return;

    const int n  = lane & 15;     // N (= t offset) this lane holds in B/C/D
    const int hi = lane >> 4;

    // B (32x16 f16): element j -> K = 8*hi + j (j<8), K = 16 + 8*hi + (j-8) (j>=8), N = lane%16.
    // Live columns K<16 only; elements 8..15 are padding zeros.
    v16h bmat;
#pragma unroll
    for (int j = 0; j < 16; ++j) {
        const int k = (j < 8) ? (hi * 8 + j) : (16 + hi * 8 + (j - 8));
        bmat[j] = (_Float16)((k < 16 && k <= n) ? (float)(n - k + 1) : 0.0f);
    }

    const size_t arow = (size_t)(rowbase + n) * Tt;  // A row this lane loads
    v8f cacc = {};                                   // carry tile (C), zero-init

    for (int t0 = 0; t0 < Tt; t0 += 16) {
        // A (16x32 f16): lanes<16 elements 0..7 = K0..7, lanes>=16 = K8..15;
        // elements 8..15 (K16..31) zeroed.  16B-granule contiguous loads.
        const float* sp = spikes + arow + t0 + hi * 8;
        const float4 f0 = *(const float4*)(sp + 0);
        const float4 f1 = *(const float4*)(sp + 4);
        v16h a = {};
        a[0] = (_Float16)f0.x; a[1] = (_Float16)f0.y;
        a[2] = (_Float16)f0.z; a[3] = (_Float16)f0.w;
        a[4] = (_Float16)f1.x; a[5] = (_Float16)f1.y;
        a[6] = (_Float16)f1.z; a[7] = (_Float16)f1.w;

        // D = A x B + C   (v_wmma_f32_16x16x32_f16)
        v8f d = __builtin_amdgcn_wmma_f32_16x16x32_f16(
            false, a, false, bmat, (short)0, cacc, false, false);

        // Stores (64B contiguous per half-wave per VGPR) + carry extraction.
        v8f cn;
#pragma unroll
        for (int r = 0; r < 8; ++r) {
            const int    row = rowbase + r + hi * 8;   // M = r + 8*hi
            const size_t idx = (size_t)row * Tt + t0 + n;
            const float  z   = d[r];
            z_out[idx] = z;
            out[idx]   = (z == 1.0f) ? 1.0f : 0.0f;
            const float c2v = __shfl(d[r], hi ? 31 : 15, 32);  // z at chunk end
            const float c14 = __shfl(d[r], hi ? 30 : 14, 32);
            const float c1v = c2v - c14;                       // spike cumsum
            cn[r] = (float)(n + 1) * c1v + c2v;
        }
        cacc = cn;
    }
}

extern "C" void kernel_launch(void* const* d_in, const int* in_sizes, int n_in,
                              void* d_out, int out_size, void* d_ws, size_t ws_size,
                              hipStream_t stream) {
    (void)in_sizes; (void)n_in; (void)out_size; (void)d_ws; (void)ws_size;
    const float* current = (const float*)d_in[0];
    const float* beta    = (const float*)d_in[1];
    const float* v_init  = (const float*)d_in[2];
    const float* v_th    = (const float*)d_in[3];
    float* outp = (float*)d_out;   // [out | z | membrane], each NTOT floats

    // 16384 waves (one per row), 8 waves/block
    snn_scan_kernel<<<NROWS / 8, 256, 0, stream>>>(
        current, beta, v_init, v_th, outp /*spike scratch*/, outp + 2 * NTOT);

    // 1024 waves (one per 16-row group), 8 waves/block
    snn_phi_wmma_kernel<<<(NROWS / 16) / 8, 256, 0, stream>>>(
        outp, outp, outp + NTOT);
}
